// NystromAttention_77120432767556
// MI455X (gfx1250) — compile-verified, matrix operand order checked by reading, not run on hardware
//
#include <hip/hip_runtime.h>

// ---------------- Types & WMMA helpers (CDNA5 / gfx1250, wave32) ----------------
typedef _Float16 f16;
typedef __attribute__((ext_vector_type(16))) _Float16 v16h;
typedef __attribute__((ext_vector_type(8)))  float    v8f;

#define DEV __device__ __forceinline__
// 1 wave32 per block; min 1 wave/EU so the compiler may use the full VGPR file
// (avoids accumulator spills in 64x64-tile GEMMs).
#define WMMA_KERNEL __global__ __launch_bounds__(32, 1)

DEV v8f wmma16(v16h a, v16h b, v8f c) {
  // D = A(16x32 f16) x B(32x16 f16) + C(16x16 f32)
  return __builtin_amdgcn_wmma_f32_16x16x32_f16(
      /*neg_a=*/false, a, /*neg_b=*/false, b,
      /*c_mod=*/(short)0, c, /*reuse_a=*/false, /*reuse_b=*/false);
}

// A-style fragment from row-major f16 matrix: rows [row..row+15], K-slice [k0..k0+31].
// Per ISA 16-bit A layout: m = lane&15, kbase=(lane>>4)*8; halves 0..7 -> K=kbase+t,
// halves 8..15 -> K=16+kbase+t. Two contiguous 16B runs per lane (global_load_b128).
DEV v16h frag_f16(const f16* p, int row, int ld, int k0) {
  int lane = threadIdx.x & 31;
  const f16* base = p + (size_t)(row + (lane & 15)) * ld + k0 + ((lane >> 4) << 3);
  v16h f;
  ((float4*)&f)[0] = *(const float4*)(base);
  ((float4*)&f)[1] = *(const float4*)(base + 16);
  return f;
}

// C/D fragment element position: row = i + 8*(lane>=16), col = lane&15
DEV int cfrag_row(int i) { return i + (((threadIdx.x & 31) >> 4) << 3); }
DEV int cfrag_col()      { return threadIdx.x & 15; }

// ---------------- Problem constants ----------------
#define BB   2
#define NN   8192
#define DIMM 512
#define HH   8
#define DH   64
#define MM   256
#define SEG  32
#define BH   16           // B*H
#define SCALE_Q 0.125f    // 64^-0.5

// ---------------- K0a: straight convert f32 -> f16 ----------------
__global__ void k_cvt_f16(const float* __restrict__ src, f16* __restrict__ dst, int n) {
  int idx = blockIdx.x * blockDim.x + threadIdx.x;
  if (idx < n) dst[idx] = (f16)src[idx];
}

// ---------------- K0b: transpose+convert f32 [K][C] -> f16 [C][K] ----------------
__global__ void k_transpose_cvt(const float* __restrict__ src, f16* __restrict__ dst,
                                int K, int C) {
  int idx = blockIdx.x * blockDim.x + threadIdx.x;
  if (idx >= K * C) return;
  int c = idx % C, k = idx / C;
  dst[(size_t)c * K + k] = (f16)src[idx];
}

// ---------------- K2: QKV GEMM [16384x512]@[512x1536] -> q,k,v (+vT) f16 ----------------
WMMA_KERNEL void k_qkv_gemm(const f16* __restrict__ A, const f16* __restrict__ BT,
                            f16* __restrict__ q_h, f16* __restrict__ k_h,
                            f16* __restrict__ v_h, f16* __restrict__ vT_h) {
  int rt = blockIdx.x;   // row tile: rows rt*64, 256 tiles
  int ct = blockIdx.y;   // col tile: cols ct*64, 24 tiles
  v8f c[4][4] = {};
  for (int ks = 0; ks < 16; ++ks) {
    int k0 = ks * 32;
    v16h a[4];
#pragma unroll
    for (int i = 0; i < 4; ++i) a[i] = frag_f16(A, rt * 64 + i * 16, DIMM, k0);
#pragma unroll
    for (int j = 0; j < 4; ++j) {
      v16h b = frag_f16(BT, ct * 64 + j * 16, DIMM, k0);  // one B frag live
#pragma unroll
      for (int i = 0; i < 4; ++i) c[i][j] = wmma16(a[i], b, c[i][j]);
    }
  }
  int n = cfrag_col();
  for (int i4 = 0; i4 < 4; ++i4)
    for (int j4 = 0; j4 < 4; ++j4)
#pragma unroll
      for (int e = 0; e < 8; ++e) {
        int R = rt * 64 + i4 * 16 + cfrag_row(e);
        int C = ct * 64 + j4 * 16 + n;
        int which = C / DIMM;           // 0=q 1=k 2=v
        int cc = C % DIMM;
        int head = cc / DH, d = cc % DH;
        int b0 = R >> 13, pos = R & (NN - 1);
        float v = c[i4][j4][e];
        size_t off = ((size_t)(b0 * HH + head) * NN + pos) * DH + d;
        if (which == 0)      q_h[off] = (f16)(v * SCALE_Q);
        else if (which == 1) k_h[off] = (f16)v;
        else {
          v_h[off] = (f16)v;
          vT_h[((size_t)(b0 * HH + head) * DH + d) * NN + pos] = (f16)v;
        }
      }
}

// ---------------- K3: landmarks = segment mean over 32 positions ----------------
__global__ void k_landmarks(const f16* __restrict__ q_h, const f16* __restrict__ k_h,
                            f16* __restrict__ ql, f16* __restrict__ kl) {
  const int per = BH * MM * DH;
  int idx = blockIdx.x * blockDim.x + threadIdx.x;
  if (idx >= 2 * per) return;
  int sel = idx / per;
  int r = idx % per;
  int d = r & 63, mi = (r >> 6) & 255, bh = r >> 14;
  const f16* src = sel ? k_h : q_h;
  f16* dst = sel ? kl : ql;
  float s = 0.f;
  for (int t = 0; t < SEG; ++t)
    s += (float)src[((size_t)bh * NN + mi * SEG + t) * DH + d];
  dst[((size_t)bh * MM + mi) * DH + d] = (f16)(s * (1.f / SEG));
}

// ---------------- K4: attn2 = softmax(q_l @ k_l^T) + column sums ----------------
WMMA_KERNEL void k_attn2(const f16* __restrict__ ql, const f16* __restrict__ kl,
                         f16* __restrict__ X_h, float* __restrict__ colsum) {
  __shared__ float S[16][256];
  int mt = blockIdx.x, bh = blockIdx.y;
  const f16* qlb = ql + (size_t)bh * MM * DH;
  const f16* klb = kl + (size_t)bh * MM * DH;
  v16h a0 = frag_f16(qlb, mt * 16, DH, 0);
  v16h a1 = frag_f16(qlb, mt * 16, DH, 32);
  int n = cfrag_col();
  for (int j = 0; j < 16; ++j) {
    v8f s = {};
    s = wmma16(a0, frag_f16(klb, j * 16, DH, 0), s);
    s = wmma16(a1, frag_f16(klb, j * 16, DH, 32), s);
#pragma unroll
    for (int e = 0; e < 8; ++e) S[cfrag_row(e)][j * 16 + n] = s[e];
  }
  __syncthreads();
  int lane = threadIdx.x;
  if (lane < 16) {
    float mx = -1e30f;
    for (int c2 = 0; c2 < 256; ++c2) mx = fmaxf(mx, S[lane][c2]);
    float sum = 0.f;
    for (int c2 = 0; c2 < 256; ++c2) { float e2 = __expf(S[lane][c2] - mx); S[lane][c2] = e2; sum += e2; }
    float inv = 1.f / sum;
    int row = mt * 16 + lane;
    for (int c2 = 0; c2 < 256; ++c2) {
      float vv = S[lane][c2] * inv;
      X_h[((size_t)bh * MM + row) * MM + c2] = (f16)vv;
      atomicAdd(&colsum[bh * MM + c2], vv);
    }
  }
}

// ---------------- scale = rowmax(=1) * max colsum (global over b,h) ----------------
__global__ void k_scale(const float* __restrict__ colsum, float* __restrict__ scale) {
  __shared__ float red[256];
  float mx = 0.f;
  for (int i = threadIdx.x; i < BH * MM; i += 256) mx = fmaxf(mx, colsum[i]);
  red[threadIdx.x] = mx;
  __syncthreads();
  for (int s = 128; s > 0; s >>= 1) {
    if (threadIdx.x < (unsigned)s) red[threadIdx.x] = fmaxf(red[threadIdx.x], red[threadIdx.x + s]);
    __syncthreads();
  }
  if (threadIdx.x == 0) scale[0] = red[0];
}

// ---------------- init Z = X^T / scale (and ZT) ----------------
__global__ void k_initz(const f16* __restrict__ X_h, const float* __restrict__ scale,
                        f16* __restrict__ Z, f16* __restrict__ ZT) {
  int idx = blockIdx.x * blockDim.x + threadIdx.x;
  if (idx >= BH * MM * MM) return;
  int j = idx & 255, i = (idx >> 8) & 255, bh = idx >> 16;
  float inv = 1.f / scale[0];
  float xij = (float)X_h[((size_t)bh * MM + i) * MM + j];
  ZT[((size_t)bh * MM + i) * MM + j] = (f16)(xij * inv);  // ZT[i][j] = Z[j][i] = X[i][j]/s
  Z [((size_t)bh * MM + j) * MM + i] = (f16)(xij * inv);
}

// ------- generic batched GEMM: OUT = alpha*(A[256x256] @ B) + beta*D, B given as BT -------
WMMA_KERNEL void k_gemm256(const f16* __restrict__ A, const f16* __restrict__ BT,
                           const f16* __restrict__ Dm, f16* __restrict__ OUT,
                           f16* __restrict__ OUTT, int Ncols, float alpha, float beta) {
  int rt = blockIdx.x, ct = blockIdx.y, bh = blockIdx.z;
  const f16* Ab = A + (size_t)bh * MM * MM;
  const f16* Bb = BT + (size_t)bh * Ncols * MM;
  v8f c[4][4] = {};
  for (int ks = 0; ks < 8; ++ks) {
    int k0 = ks * 32;
    v16h a[4];
#pragma unroll
    for (int i = 0; i < 4; ++i) a[i] = frag_f16(Ab, rt * 64 + i * 16, MM, k0);
#pragma unroll
    for (int j = 0; j < 4; ++j) {
      v16h b = frag_f16(Bb, ct * 64 + j * 16, MM, k0);
#pragma unroll
      for (int i = 0; i < 4; ++i) c[i][j] = wmma16(a[i], b, c[i][j]);
    }
  }
  int n = cfrag_col();
  for (int i4 = 0; i4 < 4; ++i4)
    for (int j4 = 0; j4 < 4; ++j4)
#pragma unroll
      for (int e = 0; e < 8; ++e) {
        int r = rt * 64 + i4 * 16 + cfrag_row(e);
        int cc = ct * 64 + j4 * 16 + n;
        float v = alpha * c[i4][j4][e];
        if (Dm) v += beta * (float)Dm[((size_t)bh * MM + r) * Ncols + cc];
        if (OUT)  OUT [((size_t)bh * MM + r) * Ncols + cc] = (f16)v;
        if (OUTT) OUTT[((size_t)bh * Ncols + cc) * MM + r] = (f16)v;
      }
}

// ------- K5: flash-style  (softmax(q_l @ k^T) @ v)^T  -> A3VT [bh][64][256] -------
WMMA_KERNEL void k_attn3v(const f16* __restrict__ ql, const f16* __restrict__ k_h,
                          const f16* __restrict__ vT, f16* __restrict__ A3VT) {
  __shared__ float Sld[16][32];
  __shared__ f16   Pld[16][32];
  __shared__ float rmax[16], rsum[16], rcorr[16];
  int mt = blockIdx.x, bh = blockIdx.y;
  const f16* qlb = ql  + (size_t)bh * MM * DH;
  const f16* kb  = k_h + (size_t)bh * NN * DH;
  const f16* vtb = vT  + (size_t)bh * DH * NN;
  v16h a0 = frag_f16(qlb, mt * 16, DH, 0);
  v16h a1 = frag_f16(qlb, mt * 16, DH, 32);
  v8f o[4] = {};
  int lane = threadIdx.x;
  if (lane < 16) { rmax[lane] = -1e30f; rsum[lane] = 0.f; }
  __syncthreads();
  int n = cfrag_col();
  for (int cb = 0; cb < NN; cb += 32) {
#pragma unroll
    for (int half = 0; half < 2; ++half) {
      v8f s = {};
      s = wmma16(a0, frag_f16(kb, cb + half * 16, DH, 0), s);
      s = wmma16(a1, frag_f16(kb, cb + half * 16, DH, 32), s);
#pragma unroll
      for (int e = 0; e < 8; ++e) Sld[cfrag_row(e)][half * 16 + n] = s[e];
    }
    __syncthreads();
    if (lane < 16) {
      float mx = rmax[lane];
      for (int jj = 0; jj < 32; ++jj) mx = fmaxf(mx, Sld[lane][jj]);
      float corr = __expf(rmax[lane] - mx);
      float sum = rsum[lane] * corr;
      for (int jj = 0; jj < 32; ++jj) {
        float e2 = __expf(Sld[lane][jj] - mx);
        Pld[lane][jj] = (f16)e2;
        sum += e2;
      }
      rmax[lane] = mx; rsum[lane] = sum; rcorr[lane] = corr;
    }
    __syncthreads();
#pragma unroll
    for (int j = 0; j < 4; ++j)
#pragma unroll
      for (int e = 0; e < 8; ++e) o[j][e] *= rcorr[cfrag_row(e)];
    v16h pa = frag_f16(&Pld[0][0], 0, 32, 0);
#pragma unroll
    for (int j = 0; j < 4; ++j)
      o[j] = wmma16(pa, frag_f16(vtb + cb, j * 16, NN, 0), o[j]);
    __syncthreads();
  }
#pragma unroll
  for (int j = 0; j < 4; ++j)
#pragma unroll
    for (int e = 0; e < 8; ++e) {
      int row = cfrag_row(e);
      int dim = j * 16 + n;
      A3VT[((size_t)bh * DH + dim) * MM + mt * 16 + row] = (f16)(o[j][e] / rsum[row]);
    }
}

// ------- K6: out = softmax(q @ k_l^T) @ W2  (fused, never materialize attn1) -------
WMMA_KERNEL void k_attn1_out(const f16* __restrict__ q_h, const f16* __restrict__ kl,
                             const f16* __restrict__ W2T, f16* __restrict__ attnout_h) {
  __shared__ float S[16][256];
  __shared__ f16   P[16][256];
  __shared__ float rinv[16];
  int nt = blockIdx.x, bh = blockIdx.y;
  const f16* qb  = q_h + (size_t)bh * NN * DH;
  const f16* klb = kl  + (size_t)bh * MM * DH;
  v16h a0 = frag_f16(qb, nt * 16, DH, 0);
  v16h a1 = frag_f16(qb, nt * 16, DH, 32);
  int n = cfrag_col();
  for (int j = 0; j < 16; ++j) {
    v8f s = {};
    s = wmma16(a0, frag_f16(klb, j * 16, DH, 0), s);
    s = wmma16(a1, frag_f16(klb, j * 16, DH, 32), s);
#pragma unroll
    for (int e = 0; e < 8; ++e) S[cfrag_row(e)][j * 16 + n] = s[e];
  }
  __syncthreads();
  int lane = threadIdx.x;
  if (lane < 16) {
    float mx = -1e30f;
    for (int c2 = 0; c2 < 256; ++c2) mx = fmaxf(mx, S[lane][c2]);
    float sum = 0.f;
    for (int c2 = 0; c2 < 256; ++c2) { float e2 = __expf(S[lane][c2] - mx); P[lane][c2] = (f16)e2; sum += e2; }
    rinv[lane] = 1.f / sum;
  }
  __syncthreads();
  v8f o[4] = {};
  for (int ks = 0; ks < 8; ++ks) {
    v16h pa = frag_f16(&P[0][0], 0, 256, ks * 32);
#pragma unroll
    for (int j = 0; j < 4; ++j)
      o[j] = wmma16(pa, frag_f16(W2T + (size_t)bh * DH * MM, j * 16, MM, ks * 32), o[j]);
  }
  int b0 = bh >> 3, h = bh & 7;
#pragma unroll
  for (int j = 0; j < 4; ++j)
#pragma unroll
    for (int e = 0; e < 8; ++e) {
      int row = cfrag_row(e);
      int pos = nt * 16 + row;
      int dim = j * 16 + n;
      attnout_h[((size_t)b0 * NN + pos) * DIMM + h * DH + dim] = (f16)(o[j][e] * rinv[row]);
    }
}

// ------- K7: grouped conv residual (kernel (33,1) over (h,n), groups=8) -------
__global__ void k_conv_add(const f16* __restrict__ v_h, const float* __restrict__ cw,
                           f16* __restrict__ attnout_h) {
  __shared__ f16 vloc[512];
  int bn = blockIdx.x;
  int b0 = bn >> 13, nn2 = bn & (NN - 1);
  int t = threadIdx.x;                 // 512 threads = (h, o)
  int hp = t >> 6, dd = t & 63;
  vloc[t] = v_h[(((size_t)b0 * HH + hp) * NN + nn2) * DH + dd];
  __syncthreads();
  int h = t >> 6, o = t & 63;
  int g = o >> 3;
  float acc = 0.f;
  for (int h2 = 0; h2 < 8; ++h2) {
    int kh = h2 - h + 16;              // always in [9,24] subset of [0,33)
#pragma unroll
    for (int i = 0; i < 8; ++i)
      acc += (float)vloc[h2 * 64 + g * 8 + i] * cw[((size_t)kh * 8 + i) * 64 + o];
  }
  size_t idx = ((size_t)b0 * NN + nn2) * DIMM + h * DH + o;
  attnout_h[idx] = (f16)((float)attnout_h[idx] + acc);
}

// ------- K8: final GEMM [16384x512]@[512x512] + bias -> f32 out -------
WMMA_KERNEL void k_out_gemm(const f16* __restrict__ A, const f16* __restrict__ BT,
                            const float* __restrict__ bias, float* __restrict__ out) {
  int rt = blockIdx.x, ct = blockIdx.y;
  v8f c[4][4] = {};
  for (int ks = 0; ks < 16; ++ks) {
    int k0 = ks * 32;
    v16h a[4];
#pragma unroll
    for (int i = 0; i < 4; ++i) a[i] = frag_f16(A, rt * 64 + i * 16, DIMM, k0);
#pragma unroll
    for (int j = 0; j < 4; ++j) {
      v16h b = frag_f16(BT, ct * 64 + j * 16, DIMM, k0);
#pragma unroll
      for (int i = 0; i < 4; ++i) c[i][j] = wmma16(a[i], b, c[i][j]);
    }
  }
  int n = cfrag_col();
  for (int i4 = 0; i4 < 4; ++i4)
    for (int j4 = 0; j4 < 4; ++j4)
#pragma unroll
      for (int e = 0; e < 8; ++e) {
        int R = rt * 64 + i4 * 16 + cfrag_row(e);
        int C = ct * 64 + j4 * 16 + n;
        out[(size_t)R * DIMM + C] = c[i4][j4][e] + bias[C];
      }
}

// ---------------- host orchestration ----------------
extern "C" void kernel_launch(void* const* d_in, const int* in_sizes, int n_in,
                              void* d_out, int out_size, void* d_ws, size_t ws_size,
                              hipStream_t stream) {
  const float* inputs = (const float*)d_in[0];   // [2,8192,512]
  const float* Wqkv   = (const float*)d_in[1];   // [512,1536]
  const float* Wout   = (const float*)d_in[2];   // [512,512]
  const float* bout   = (const float*)d_in[3];   // [512]
  const float* conv_w = (const float*)d_in[4];   // [33,1,8,64]
  float* out = (float*)d_out;

  char* p = (char*)d_ws;
  size_t off = 0;
  auto alloc = [&](size_t bytes) -> char* {
    char* r = p + off;
    off = (off + bytes + 255) & ~(size_t)255;
    return r;
  };
  const size_t S_QKV = (size_t)BH * NN * DH * sizeof(f16);   // 16 MB each
  const size_t S_MAT = (size_t)BH * MM * MM * sizeof(f16);   // 2 MB
  f16* in_h   = (f16*)alloc((size_t)BB * NN * DIMM * sizeof(f16));
  f16* WqkvT  = (f16*)alloc((size_t)1536 * 512 * sizeof(f16));
  f16* WoutT  = (f16*)alloc((size_t)512 * 512 * sizeof(f16));
  f16* q_h    = (f16*)alloc(S_QKV);
  f16* k_h    = (f16*)alloc(S_QKV);
  f16* v_h    = (f16*)alloc(S_QKV);
  f16* vT_h   = (f16*)alloc(S_QKV);
  f16* ql_h   = (f16*)alloc((size_t)BH * MM * DH * sizeof(f16));
  f16* kl_h   = (f16*)alloc((size_t)BH * MM * DH * sizeof(f16));
  f16* X_h    = (f16*)alloc(S_MAT);
  float* colsum = (float*)alloc((size_t)BH * MM * sizeof(float));
  float* scale  = (float*)alloc(256);
  f16* W_h    = (f16*)alloc(S_MAT);
  f16* WT_h   = (f16*)alloc(S_MAT);
  f16* PT_h   = (f16*)alloc(S_MAT);
  f16* QT_h   = (f16*)alloc(S_MAT);
  f16* Za     = (f16*)alloc(S_MAT);
  f16* ZaT    = (f16*)alloc(S_MAT);
  f16* Zb     = (f16*)alloc(S_MAT);
  f16* ZbT    = (f16*)alloc(S_MAT);
  f16* A3VT   = (f16*)alloc((size_t)BH * DH * MM * sizeof(f16));
  f16* W2T    = (f16*)alloc((size_t)BH * DH * MM * sizeof(f16));
  f16* attnout_h = (f16*)alloc((size_t)BB * NN * DIMM * sizeof(f16));
  (void)ws_size; (void)in_sizes; (void)n_in; (void)out_size;

  // Conversions: activations straight, weights transposed (WMMA B operands)
  k_cvt_f16<<<(BB * NN * DIMM + 255) / 256, 256, 0, stream>>>(inputs, in_h, BB * NN * DIMM);
  k_transpose_cvt<<<(512 * 1536 + 255) / 256, 256, 0, stream>>>(Wqkv, WqkvT, 512, 1536);
  k_transpose_cvt<<<(512 * 512 + 255) / 256, 256, 0, stream>>>(Wout, WoutT, 512, 512);

  // QKV projection
  k_qkv_gemm<<<dim3(256, 24), 32, 0, stream>>>(in_h, WqkvT, q_h, k_h, v_h, vT_h);

  // Landmarks
  k_landmarks<<<(2 * BH * MM * DH + 255) / 256, 256, 0, stream>>>(q_h, k_h, ql_h, kl_h);

  // attn3 @ v (flash style), independent of attn2 chain
  k_attn3v<<<dim3(MM / 16, BH), 32, 0, stream>>>(ql_h, k_h, vT_h, A3VT);

  // attn2 softmax + colsums -> scale -> Z0
  hipMemsetAsync(colsum, 0, (size_t)BH * MM * sizeof(float), stream);
  k_attn2<<<dim3(MM / 16, BH), 32, 0, stream>>>(ql_h, kl_h, X_h, colsum);
  k_scale<<<1, 256, 0, stream>>>(colsum, scale);
  k_initz<<<(BH * MM * MM + 255) / 256, 256, 0, stream>>>(X_h, scale, Za, ZaT);

  // pinv iterations: W = X@Z; P = 7W - W@W; Q = 15W - W@P; Z' = 3.25Z - 0.25 Z@Q
  f16 *Zc = Za, *ZcT = ZaT, *Zn = Zb, *ZnT = ZbT;
  for (int it = 0; it < 6; ++it) {
    k_gemm256<<<dim3(4, 4, BH), 32, 0, stream>>>(X_h, ZcT, nullptr, W_h, WT_h, 256, 1.f, 0.f);
    k_gemm256<<<dim3(4, 4, BH), 32, 0, stream>>>(W_h, WT_h, W_h, nullptr, PT_h, 256, -1.f, 7.f);
    k_gemm256<<<dim3(4, 4, BH), 32, 0, stream>>>(W_h, PT_h, W_h, nullptr, QT_h, 256, -1.f, 15.f);
    k_gemm256<<<dim3(4, 4, BH), 32, 0, stream>>>(Zc, QT_h, Zc, Zn, ZnT, 256, -0.25f, 3.25f);
    f16* t;
    t = Zc; Zc = Zn; Zn = t;
    t = ZcT; ZcT = ZnT; ZnT = t;
  }

  // W2 = pinv(attn2) @ (attn3 @ v)   [256x64 per (b,h)], keep transpose only
  k_gemm256<<<dim3(4, 1, BH), 32, 0, stream>>>(Zc, A3VT, nullptr, nullptr, W2T, 64, 1.f, 0.f);

  // attn1 softmax fused with @W2 -> attnout
  k_attn1_out<<<dim3(NN / 16, BH), 32, 0, stream>>>(q_h, kl_h, W2T, attnout_h);

  // conv residual add
  k_conv_add<<<BB * NN, 512, 0, stream>>>(v_h, conv_w, attnout_h);

  // output projection + bias
  k_out_gemm<<<dim3(256, 8), 32, 0, stream>>>(attnout_h, WoutT, bout, out);
}
